// NequIP_16561393893486
// MI455X (gfx1250) — matrix operand order repaired and checked by reading, not compiled, not run on hardware
//
#include <hip/hip_runtime.h>

// ---------------- problem constants ----------------
#define N_NODES 8192
#define N_EDGES 131072
#define CC      64
#define HH      64
#define NRAD    8
#define NLAYERS 2
#define NPATHS  15
#define RMAX_INV 0.2f
#define INV_SQRT_NN 0.25f   // 1/sqrt(16)

// CG constants
#define RS2  0.70710678f   // 1/sqrt(2)
#define RT6  0.40824829f   // 1/sqrt(6)
#define SQ3  1.73205081f
#define RS3  0.57735027f   // 1/sqrt(3)
#define RS5  0.44721360f   // 1/sqrt(5)
#define RS10 0.31622777f   // 1/sqrt(10)
#define C35  0.77459667f   // sqrt(3/5)
#define C222 0.65465367f   // sqrt(3/7)
#define C2A  0.81649658f   // sqrt(2/3)
#define C2B  0.40824829f

typedef __attribute__((ext_vector_type(16))) _Float16 v16h;
typedef __attribute__((ext_vector_type(8)))  _Float16 v8h;
typedef __attribute__((ext_vector_type(8)))  float    v8f;

// fast rcp (v_rcp_f32, no IEEE div fixup chain)
__device__ __forceinline__ float fast_rcp(float x){ return __builtin_amdgcn_rcpf(x); }
__device__ __forceinline__ float silu_f(float x){ return x * fast_rcp(1.f + __expf(-x)); }
__device__ __forceinline__ float sigm_f(float x){ return fast_rcp(1.f + __expf(-x)); }

#define WMMA_F16(A,B,C) __builtin_amdgcn_wmma_f32_16x16x32_f16(false, (A), false, (B), (short)0, (C), false, false)

// A fragment: 16x32 f16 tile from row-major LDS [16][64], K-block kb (contiguous k = 16*(lane/16) + e)
__device__ __forceinline__ v16h load_a_frag(const _Float16* h, int lane, int kb){
  int m = lane & 15;
  int kbase = kb*32 + ((lane & 16) ? 16 : 0);
  const _Float16* p = h + m*64 + kbase;
  v8h lo = *(const v8h*)(p);
  v8h hi = *(const v8h*)(p + 8);
  v16h a;
#pragma unroll
  for (int e = 0; e < 8; ++e){ a[e] = lo[e]; a[e+8] = hi[e]; }
  return a;
}

// B fragment: prepacked [frag][lane][16] f16, 32B coalesced load per lane
__device__ __forceinline__ v16h load_b_frag(const _Float16* pack, int frag, int lane){
  return *(const v16h*)(pack + ((size_t)frag*32 + lane)*16);
}

// ---------------- geometry kernel ----------------
__global__ __launch_bounds__(256) void geom_kernel(
    const float* __restrict__ pos, const int* __restrict__ snd, const int* __restrict__ rcv,
    float* __restrict__ gsh1, float* __restrict__ gsh2,
    float* __restrict__ gbasis, float* __restrict__ gcut)
{
  int e = blockIdx.x*256 + threadIdx.x;
  if (e >= N_EDGES) return;
  int s = snd[e], r = rcv[e];
  float rx = (pos[r*3+0]-pos[s*3+0])*RMAX_INV;
  float ry = (pos[r*3+1]-pos[s*3+1])*RMAX_INV;
  float rz = (pos[r*3+2]-pos[s*3+2])*RMAX_INV;
  float d  = sqrtf(rx*rx+ry*ry+rz*rz);
  float iv = fast_rcp(fmaxf(d, 1e-6f));
  float x = rx*iv, y = ry*iv, z = rz*iv;
  gsh1[e*3+0] = 1.7320508f*y; gsh1[e*3+1] = 1.7320508f*z; gsh1[e*3+2] = 1.7320508f*x;
  gsh2[e*5+0] = 3.8729833f*x*y;
  gsh2[e*5+1] = 3.8729833f*y*z;
  gsh2[e*5+2] = 1.1180340f*(3.f*z*z - 1.f);
  gsh2[e*5+3] = 3.8729833f*x*z;
  gsh2[e*5+4] = 1.9364917f*(x*x - y*y);
  float xr = fminf(fmaxf(d, 1e-4f), 1.f);
  float ixr = 1.4142136f * fast_rcp(xr);
#pragma unroll
  for (int f = 0; f < NRAD; ++f)
    gbasis[e*8+f] = __sinf((float)(f+1)*3.14159265f*xr) * ixr;
  float dc = fminf(fmaxf(d, 0.f), 1.f);
  gcut[e] = 0.5f*(__cosf(3.14159265f*dc) + 1.f);
}

// ---------------- weight packing (f32 -> f16 WMMA B-fragments) ----------------
__global__ __launch_bounds__(256) void pack_kernel(
    const float* __restrict__ w2, const float* __restrict__ w3, const float* __restrict__ ls,
    _Float16* __restrict__ w2p, _Float16* __restrict__ w3p, _Float16* __restrict__ lsp)
{
  const int W2TOT = NLAYERS*4096, W3TOT = NLAYERS*61440, LSTOT = NLAYERS*12288;
  int idx = blockIdx.x*256 + threadIdx.x;
  if (idx < W2TOT) {
    int L = idx/4096, r = idx%4096;
    int e = r&15, lane = (r>>4)&31, ct = (r>>9)&3, kb = r>>11;
    int k = kb*32 + ((lane<16)?0:16) + e, n = ct*16 + (lane&15);
    w2p[idx] = (_Float16)w2[(L*64 + k)*64 + n];
  } else if (idx < W2TOT + W3TOT) {
    int j = idx - W2TOT;
    int L = j/61440, r = j%61440;
    int e = r&15, lane = (r>>4)&31, ck = r>>9, ct = ck%60, kb = ck/60;
    int k = kb*32 + ((lane<16)?0:16) + e, n = ct*16 + (lane&15);
    w3p[j] = (_Float16)w3[(L*64 + k)*960 + n];
  } else if (idx < W2TOT + W3TOT + LSTOT) {
    int j = idx - W2TOT - W3TOT;
    int L = j/12288, r = j%12288;
    int e = r&15, lane = (r>>4)&31, ct = (r>>9)&3, kbl = r>>11, kb = kbl&1, l = kbl>>1;
    int k = kb*32 + ((lane<16)?0:16) + e, n = ct*16 + (lane&15);
    lsp[j] = (_Float16)ls[(((L*3 + l)*64) + k)*64 + n];
  }
}

// ---------------- node embedding projection ----------------
__global__ __launch_bounds__(256) void init_kernel(
    const int* __restrict__ species, const float* __restrict__ embed,
    const float* __restrict__ wproj, float* __restrict__ f0)
{
  int i = blockIdx.x*256 + threadIdx.x;
  if (i >= N_NODES*CC) return;
  int node = i >> 6, c = i & 63;
  int sp = species[node];
  float acc = 0.f;
#pragma unroll
  for (int k = 0; k < 32; ++k) acc = fmaf(embed[sp*32 + k], wproj[k*64 + c], acc);
  f0[i] = acc;
}

// ---------------- fused edge kernel: radial MLP (WMMA) + messages + scatter ----------------
__global__ __launch_bounds__(128) void edge_kernel(
    const int* __restrict__ snd, const int* __restrict__ rcv,
    const float* __restrict__ gsh1, const float* __restrict__ gsh2,
    const float* __restrict__ gbasis, const float* __restrict__ gcut,
    const float* __restrict__ f0, const float* __restrict__ f1, const float* __restrict__ f2,
    const float* __restrict__ w1L, const float* __restrict__ b1L,
    const _Float16* __restrict__ w2p, const float* __restrict__ b2L,
    const _Float16* __restrict__ w3p,
    float* __restrict__ agg0, float* __restrict__ agg1, float* __restrict__ agg2)
{
  __shared__ float s_sh1[16][3];
  __shared__ float s_sh2[16][5];
  __shared__ float s_basis[16][8];
  __shared__ float s_cut[16];
  __shared__ __align__(32) _Float16 s_h1[16][64];
  __shared__ __align__(32) _Float16 s_h2[16][64];
  __shared__ __align__(32) _Float16 s_w[16][960];

  const int tid = threadIdx.x;
  const int e0  = blockIdx.x * 16;

  for (int i = tid; i < 16*3; i += 128) s_sh1[i/3][i%3] = gsh1[e0*3 + i];
  for (int i = tid; i < 16*5; i += 128) s_sh2[i/5][i%5] = gsh2[e0*5 + i];
  for (int i = tid; i < 16*8; i += 128) s_basis[i/8][i%8] = gbasis[e0*8 + i];
  for (int i = tid; i < 16;   i += 128) s_cut[i] = gcut[e0 + i];
  __syncthreads();

  // h1 = silu(basis @ W1 + b1)   (K=8, plain VALU)
  for (int i = tid; i < 1024; i += 128) {
    int m = i >> 6, o = i & 63;
    float acc = b1L[o];
#pragma unroll
    for (int k = 0; k < NRAD; ++k) acc = fmaf(s_basis[m][k], w1L[k*64 + o], acc);
    s_h1[m][o] = (_Float16)silu_f(acc);
  }
  __syncthreads();

  const int wave = tid >> 5, lane = tid & 31;
  const int n = lane & 15, mofs = (lane & 16) ? 8 : 0;

  // h2 = silu(h1 @ W2 + b2)  -- WMMA, 1 column tile per wave, K=64 chained
  {
    int ct = wave;
    v16h a0 = load_a_frag(&s_h1[0][0], lane, 0);
    v16h a1 = load_a_frag(&s_h1[0][0], lane, 1);
    v16h b0 = load_b_frag(w2p, 0*4 + ct, lane);
    v16h b1 = load_b_frag(w2p, 1*4 + ct, lane);
    v8f acc = {};
    acc = WMMA_F16(a0, b0, acc);
    acc = WMMA_F16(a1, b1, acc);
    float bias = b2L[ct*16 + n];
#pragma unroll
    for (int r = 0; r < 8; ++r)
      s_h2[r + mofs][ct*16 + n] = (_Float16)silu_f(acc[r] + bias);
  }
  __syncthreads();

  // w = (h2 @ W3) * cut -- WMMA; A hoisted to registers, 2 independent
  // accumulator chains per iteration to cover the WMMA->WMMA RAW hazard
  {
    const int ctbase = wave * 15;
    v16h a0 = load_a_frag(&s_h2[0][0], lane, 0);
    v16h a1 = load_a_frag(&s_h2[0][0], lane, 1);
    for (int j = 0; j < 14; j += 2) {
      const int ctA = ctbase + j, ctB = ctA + 1;
      v16h bA0 = load_b_frag(w3p, 0*60 + ctA, lane);
      v16h bB0 = load_b_frag(w3p, 0*60 + ctB, lane);
      v16h bA1 = load_b_frag(w3p, 1*60 + ctA, lane);
      v16h bB1 = load_b_frag(w3p, 1*60 + ctB, lane);
      v8f accA = {}, accB = {};
      accA = WMMA_F16(a0, bA0, accA);
      accB = WMMA_F16(a0, bB0, accB);
      accA = WMMA_F16(a1, bA1, accA);
      accB = WMMA_F16(a1, bB1, accB);
#pragma unroll
      for (int r = 0; r < 8; ++r) {
        int m = r + mofs;
        float cm = s_cut[m];
        s_w[m][ctA*16 + n] = (_Float16)(accA[r] * cm);
        s_w[m][ctB*16 + n] = (_Float16)(accB[r] * cm);
      }
    }
    { // tail tile
      const int ct = ctbase + 14;
      v16h b0 = load_b_frag(w3p, 0*60 + ct, lane);
      v16h b1 = load_b_frag(w3p, 1*60 + ct, lane);
      v8f acc = {};
      acc = WMMA_F16(a0, b0, acc);
      acc = WMMA_F16(a1, b1, acc);
#pragma unroll
      for (int r = 0; r < 8; ++r) {
        int m = r + mofs;
        s_w[m][ct*16 + n] = (_Float16)(acc[r] * s_cut[m]);
      }
    }
  }
  __syncthreads();

  // ---- message phase: one edge per 8 threads, 8 channels per thread ----
  const int el = tid >> 3;
  const int edge = e0 + el;
  const int s = snd[edge];
  const int r = rcv[edge];
  float a1[3], b2v[5];
#pragma unroll
  for (int i = 0; i < 3; ++i) a1[i] = s_sh1[el][i];
#pragma unroll
  for (int k = 0; k < 5; ++k) b2v[k] = s_sh2[el][k];

  for (int q = 0; q < 8; ++q) {
    int c = ((tid & 7) << 3) + q;
    float x0 = f0[s*64 + c];
    float x1[3], x2[5];
#pragma unroll
    for (int i = 0; i < 3; ++i) x1[i] = f1[(s*64 + c)*3 + i];
#pragma unroll
    for (int k = 0; k < 5; ++k) x2[k] = f2[(s*64 + c)*5 + k];
    float wq[NPATHS];
#pragma unroll
    for (int p = 0; p < NPATHS; ++p) wq[p] = (float)s_w[el][p*64 + c];

    float m0 = 0.f, m1[3] = {0.f,0.f,0.f}, m2[5] = {0.f,0.f,0.f,0.f,0.f};

    // P0 (0,0,0)
    m0 += wq[0]*x0;
    // P1 (0,1,1), P2 (0,2,2)
#pragma unroll
    for (int k = 0; k < 3; ++k) m1[k] += wq[1]*x0*a1[k];
#pragma unroll
    for (int k = 0; k < 5; ++k) m2[k] += wq[2]*x0*b2v[k];
    // P3 (1,0,1)
#pragma unroll
    for (int k = 0; k < 3; ++k) m1[k] += wq[3]*x1[k];
    // P4 (1,1,0): dot/sqrt3
    m0 += wq[4]*RS3*(x1[0]*a1[0] + x1[1]*a1[1] + x1[2]*a1[2]);
    // P5 (1,1,1): eps/sqrt2 (cyclic in (y,z,x))
    m1[0] += wq[5]*RS2*(x1[1]*a1[2] - x1[2]*a1[1]);
    m1[1] += wq[5]*RS2*(x1[2]*a1[0] - x1[0]*a1[2]);
    m1[2] += wq[5]*RS2*(x1[0]*a1[1] - x1[1]*a1[0]);
    // P6 (1,1,2): symmetric traceless
    m2[0] += wq[6]*RS2*(x1[2]*a1[0] + x1[0]*a1[2]);
    m2[1] += wq[6]*RS2*(x1[0]*a1[1] + x1[1]*a1[0]);
    m2[2] += wq[6]*(C2A*x1[1]*a1[1] - C2B*(x1[0]*a1[0] + x1[2]*a1[2]));
    m2[3] += wq[6]*RS2*(x1[2]*a1[1] + x1[1]*a1[2]);
    m2[4] += wq[6]*RS2*(x1[2]*a1[2] - x1[0]*a1[0]);
    // P7 (1,2,1): M(sh2) . v
    {
      float vx = x1[2], vy = x1[0], vz = x1[1];
      float Mx = (-RT6*b2v[2] + RS2*b2v[4])*vx + RS2*b2v[0]*vy + RS2*b2v[3]*vz;
      float My =  RS2*b2v[0]*vx + (-RT6*b2v[2] - RS2*b2v[4])*vy + RS2*b2v[1]*vz;
      float Mz =  RS2*b2v[3]*vx + RS2*b2v[1]*vy + 2.f*RT6*b2v[2]*vz;
      float sc = wq[7]*C35;
      m1[0] += sc*My; m1[1] += sc*Mz; m1[2] += sc*Mx;
    }
    // P8 (1,2,2): commutator [K(x1), sh2]
    {
      float vx = x1[2], vy = x1[0], vz = x1[1];
      const float* Q = b2v;
      float sc = wq[8]*RT6;
      m2[0] += sc*(-vx*Q[3] + vy*Q[1] + 2.f*vz*Q[4]);
      m2[1] += sc*(-SQ3*vx*Q[2] - vx*Q[4] - vy*Q[0] + vz*Q[3]);
      m2[2] += sc*( SQ3*vx*Q[1] - SQ3*vy*Q[3]);
      m2[3] += sc*( vx*Q[0] + SQ3*vy*Q[2] - vy*Q[4] - vz*Q[1]);
      m2[4] += sc*( vx*Q[1] + vy*Q[3] - 2.f*vz*Q[0]);
    }
    // P9 (2,0,2)
#pragma unroll
    for (int k = 0; k < 5; ++k) m2[k] += wq[9]*x2[k];
    // P10 (2,1,1): M(x2) . sh1
    {
      float vx = a1[2], vy = a1[0], vz = a1[1];
      float Mx = (-RT6*x2[2] + RS2*x2[4])*vx + RS2*x2[0]*vy + RS2*x2[3]*vz;
      float My =  RS2*x2[0]*vx + (-RT6*x2[2] - RS2*x2[4])*vy + RS2*x2[1]*vz;
      float Mz =  RS2*x2[3]*vx + RS2*x2[1]*vy + 2.f*RT6*x2[2]*vz;
      float sc = wq[10]*C35;
      m1[0] += sc*My; m1[1] += sc*Mz; m1[2] += sc*Mx;
    }
    // P11 (2,1,2): commutator [K(sh1), x2]
    {
      float vx = a1[2], vy = a1[0], vz = a1[1];
      const float* Q = x2;
      float sc = wq[11]*RT6;
      m2[0] += sc*(-vx*Q[3] + vy*Q[1] + 2.f*vz*Q[4]);
      m2[1] += sc*(-SQ3*vx*Q[2] - vx*Q[4] - vy*Q[0] + vz*Q[3]);
      m2[2] += sc*( SQ3*vx*Q[1] - SQ3*vy*Q[3]);
      m2[3] += sc*( vx*Q[0] + SQ3*vy*Q[2] - vy*Q[4] - vz*Q[1]);
      m2[4] += sc*( vx*Q[1] + vy*Q[3] - 2.f*vz*Q[0]);
    }
    // P12 (2,2,0)
    m0 += wq[12]*RS5*(x2[0]*b2v[0] + x2[1]*b2v[1] + x2[2]*b2v[2] + x2[3]*b2v[3] + x2[4]*b2v[4]);
    // P13 (2,2,1)
    {
      const float* A = x2; const float* B = b2v;
      float ox =  A[0]*B[3] + SQ3*A[1]*B[2] + A[1]*B[4] - SQ3*A[2]*B[1] - A[3]*B[0] - A[4]*B[1];
      float oy = -A[0]*B[1] + A[1]*B[0] + SQ3*A[2]*B[3] - SQ3*A[3]*B[2] + A[3]*B[4] - A[4]*B[3];
      float oz = -2.f*A[0]*B[4] - A[1]*B[3] + A[3]*B[1] + 2.f*A[4]*B[0];
      float sc = wq[13]*RS10;
      m1[0] += sc*oy; m1[1] += sc*oz; m1[2] += sc*ox;
    }
    // P14 (2,2,2): symmetric anticommutator coupling
    {
      const float* A = x2; const float* B = b2v;
      float sc = wq[14]*C222;
      m2[0] += sc*( RS2*(A[1]*B[3]+A[3]*B[1]) - 2.f*RT6*(A[0]*B[2]+A[2]*B[0]) );
      m2[1] += sc*( RS2*(A[0]*B[3]+A[3]*B[0]) + RT6*(A[1]*B[2]+A[2]*B[1]) - RS2*(A[1]*B[4]+A[4]*B[1]) );
      m2[2] += sc*( RT6*(-2.f*A[0]*B[0] + A[1]*B[1] + A[3]*B[3] - 2.f*A[4]*B[4] + 2.f*A[2]*B[2]) );
      m2[3] += sc*( RS2*(A[0]*B[1]+A[1]*B[0]) + RT6*(A[2]*B[3]+A[3]*B[2]) + RS2*(A[3]*B[4]+A[4]*B[3]) );
      m2[4] += sc*( RS2*(-A[1]*B[1] + A[3]*B[3]) - 2.f*RT6*(A[2]*B[4]+A[4]*B[2]) );
    }

    atomicAdd(&agg0[r*64 + c], m0);
#pragma unroll
    for (int i = 0; i < 3; ++i) atomicAdd(&agg1[(r*64 + c)*3 + i], m1[i]);
#pragma unroll
    for (int k = 0; k < 5; ++k) atomicAdd(&agg2[(r*64 + c)*5 + k], m2[k]);
  }
}

// ---------------- node update: lin_self (WMMA) + gates ----------------
__global__ __launch_bounds__(128) void node_kernel(
    const float* __restrict__ agg0, const float* __restrict__ agg1, const float* __restrict__ agg2,
    float* __restrict__ f0, float* __restrict__ f1, float* __restrict__ f2,
    const _Float16* __restrict__ lsp, const float* __restrict__ gwL, const float* __restrict__ gbL)
{
  __shared__ __align__(32) _Float16 s_a[9][16][64];
  __shared__ float s_y[9][16][64];
  __shared__ float s_g[16][64];
  const int tid = threadIdx.x;
  const int n0 = blockIdx.x * 16;

  for (int i = tid; i < 9*16*64; i += 128) {
    int t = i >> 10, rem = i & 1023, nl = rem >> 6, c = rem & 63;
    int node = n0 + nl;
    float v;
    if (t == 0)      v = agg0[node*64 + c];
    else if (t < 4)  v = agg1[(node*64 + c)*3 + (t - 1)];
    else             v = agg2[(node*64 + c)*5 + (t - 4)];
    s_a[t][nl][c] = (_Float16)(v * INV_SQRT_NN);
  }
  __syncthreads();

  const int wave = tid >> 5, lane = tid & 31;
  const int n = lane & 15, mofs = (lane & 16) ? 8 : 0;

#pragma unroll 3
  for (int jj = 0; jj < 9; ++jj) {
    int job = wave + jj*4;
    int t = job >> 2, ct = job & 3;
    int l = (t == 0) ? 0 : ((t < 4) ? 1 : 2);
    v16h a0 = load_a_frag(&s_a[t][0][0], lane, 0);
    v16h a1 = load_a_frag(&s_a[t][0][0], lane, 1);
    v16h b0 = load_b_frag(lsp, (l*2 + 0)*4 + ct, lane);
    v16h b1 = load_b_frag(lsp, (l*2 + 1)*4 + ct, lane);
    v8f acc = {};
    acc = WMMA_F16(a0, b0, acc);
    acc = WMMA_F16(a1, b1, acc);
#pragma unroll
    for (int r = 0; r < 8; ++r)
      s_y[t][r + mofs][ct*16 + n] = acc[r];
  }
  __syncthreads();

  // feats0 += silu(y0); capture gate input
  for (int i = tid; i < 1024; i += 128) {
    int nl = i >> 6, c = i & 63;
    int node = n0 + nl;
    float v = f0[node*64 + c] + silu_f(s_y[0][nl][c]);
    f0[node*64 + c] = v;
    s_g[nl][c] = v;
  }
  __syncthreads();

  // gates + l=1,2 updates
  for (int i = tid; i < 1024; i += 128) {
    int nl = i >> 6, d = i & 63;
    int node = n0 + nl;
    float a1 = gbL[d], a2 = gbL[64 + d];
#pragma unroll 8
    for (int c = 0; c < 64; ++c) {
      float gv = s_g[nl][c];
      a1 = fmaf(gv, gwL[c*64 + d], a1);
      a2 = fmaf(gv, gwL[4096 + c*64 + d], a2);
    }
    float g1 = sigm_f(a1), g2 = sigm_f(a2);
#pragma unroll
    for (int m = 0; m < 3; ++m)
      f1[(node*64 + d)*3 + m] += g1 * s_y[1 + m][nl][d];
#pragma unroll
    for (int m = 0; m < 5; ++m)
      f2[(node*64 + d)*5 + m] += g2 * s_y[4 + m][nl][d];
  }
}

// ---------------- output concat with 0.5^l scaling ----------------
__global__ __launch_bounds__(256) void out_kernel(
    const float* __restrict__ f0, const float* __restrict__ f1, const float* __restrict__ f2,
    float* __restrict__ out)
{
  int i = blockIdx.x*256 + threadIdx.x;
  if (i >= N_NODES*576) return;
  int node = i / 576, j = i % 576;
  float v;
  if (j < 64)        v = f0[node*64 + j];
  else if (j < 256){ int jj = j - 64;  v = 0.5f  * f1[node*192 + jj]; }
  else             { int jj = j - 256; v = 0.25f * f2[node*320 + jj]; }
  out[i] = v;
}

// ---------------- host orchestration ----------------
extern "C" void kernel_launch(void* const* d_in, const int* in_sizes, int n_in,
                              void* d_out, int out_size, void* d_ws, size_t ws_size,
                              hipStream_t stream)
{
  const float* positions = (const float*)d_in[0];
  const int*   species   = (const int*)  d_in[1];
  const int*   senders   = (const int*)  d_in[2];
  const int*   receivers = (const int*)  d_in[3];
  const float* embed     = (const float*)d_in[4];
  const float* w_proj    = (const float*)d_in[5];
  const float* mlp_w1    = (const float*)d_in[6];
  const float* mlp_b1    = (const float*)d_in[7];
  const float* mlp_w2    = (const float*)d_in[8];
  const float* mlp_b2    = (const float*)d_in[9];
  const float* mlp_w3    = (const float*)d_in[10];
  const float* lin_self  = (const float*)d_in[11];
  const float* gate_w    = (const float*)d_in[12];
  const float* gate_b    = (const float*)d_in[13];

  float* ws = (float*)d_ws;
  const size_t E = N_EDGES, N = N_NODES;
  size_t o = 0;
  float* gsh1   = ws + o; o += E*3;
  float* gsh2   = ws + o; o += E*5;
  float* gbasis = ws + o; o += E*8;
  float* gcut   = ws + o; o += E;
  float* f0     = ws + o; o += N*64;
  float* f1     = ws + o; o += N*192;
  float* f2     = ws + o; o += N*320;
  float* agg0   = ws + o; o += N*64;
  float* agg1   = ws + o; o += N*192;
  float* agg2   = ws + o; o += N*320;
  _Float16* w2p = (_Float16*)(ws + o);
  _Float16* w3p = w2p + (size_t)NLAYERS*4096;
  _Float16* lsp = w3p + (size_t)NLAYERS*61440;

  geom_kernel<<<(int)((E + 255)/256), 256, 0, stream>>>(positions, senders, receivers,
                                                        gsh1, gsh2, gbasis, gcut);
  {
    int packtot = NLAYERS*(4096 + 61440 + 12288);
    pack_kernel<<<(packtot + 255)/256, 256, 0, stream>>>(mlp_w2, mlp_w3, lin_self, w2p, w3p, lsp);
  }
  init_kernel<<<(int)((N*64 + 255)/256), 256, 0, stream>>>(species, embed, w_proj, f0);
  hipMemsetAsync(f1, 0, (size_t)N*(192 + 320)*sizeof(float), stream); // f1,f2 contiguous

  for (int L = 0; L < NLAYERS; ++L) {
    hipMemsetAsync(agg0, 0, (size_t)N*576*sizeof(float), stream); // agg0..2 contiguous
    edge_kernel<<<(int)(E/16), 128, 0, stream>>>(
        senders, receivers, gsh1, gsh2, gbasis, gcut, f0, f1, f2,
        mlp_w1 + (size_t)L*NRAD*HH, mlp_b1 + (size_t)L*HH,
        w2p + (size_t)L*4096, mlp_b2 + (size_t)L*HH,
        w3p + (size_t)L*61440,
        agg0, agg1, agg2);
    node_kernel<<<(int)(N/16), 128, 0, stream>>>(
        agg0, agg1, agg2, f0, f1, f2,
        lsp + (size_t)L*12288, gate_w + (size_t)L*2*4096, gate_b + (size_t)L*128);
  }

  out_kernel<<<(int)((N*576 + 255)/256), 256, 0, stream>>>(f0, f1, f2, (float*)d_out);
}
